// N_Re3_67156108640926
// MI455X (gfx1250) — matrix operand order repaired and checked by reading, not compiled, not run on hardware
//
#include <hip/hip_runtime.h>
#include <hip/hip_bf16.h>
#include <math.h>

// ---------------------------------------------------------------------------
// MI455X (gfx1250) implementation: bf16 WMMA for all large GEMMs, fp32 accum.
// Weights pre-swizzled to WMMA-fragment-major order for coalesced L2 access.
// ---------------------------------------------------------------------------

typedef __attribute__((ext_vector_type(16))) __bf16 bf16x16;
typedef __attribute__((ext_vector_type(8)))  __bf16 bf16x8;
typedef __attribute__((ext_vector_type(8)))  float  floatx8;

#define B_SZ   1024
#define T_SZ   256
#define FEATN  32
#define CCH    128
#define HIDN   512
#define NGATE  2048   // 4*HID
#define KIH    160    // 128 tcn + 32 sr_c

#define WMMA_BF16(a,b,c) \
  __builtin_amdgcn_wmma_f32_16x16x32_bf16(false,(a),false,(b),(short)0,(c),false,false)

__device__ __forceinline__ float sigf(float x) { return 1.0f / (1.0f + expf(-x)); }

__device__ __forceinline__ floatx8 splat8(float s) {
  floatx8 v; for (int i = 0; i < 8; i++) v[i] = s; return v;
}
__device__ __forceinline__ bf16x16 zero_af() {
  bf16x16 v; for (int i = 0; i < 16; i++) v[i] = (__bf16)0.0f; return v;
}

// A fragment (16x32 bf16, MxK) from a row-major K-contiguous row:
// lane m=lane&15, half=lane>>4.
// Elements e0..7  <-> K = half*8 + 0..7      (contiguous 16B)
// Elements e8..15 <-> K = 16 + half*8 + 0..7 (contiguous 16B)
__device__ __forceinline__ bf16x16 load_a(const __bf16* row, int half) {
  bf16x16 a;
  ((bf16x8*)&a)[0] = *(const bf16x8*)(row + half * 8);
  ((bf16x8*)&a)[1] = *(const bf16x8*)(row + 16 + half * 8);
  return a;
}

// B fragment (32x16 bf16, KxN) from fragment-major weight storage:
// block (kc,nt) is 1KB contiguous; lane holds 16 K values of column n=lane&15 at +lane*32B.
// One fully-coalesced 1KB load per wave.
__device__ __forceinline__ bf16x16 load_b_frag(const __bf16* base, int NT,
                                               int kc, int nt, int lane) {
  const __bf16* p = base + (((size_t)kc * NT + nt) * 32 + lane) * 16;
  bf16x16 b;
  ((bf16x8*)&b)[0] = *(const bf16x8*)(p);
  ((bf16x8*)&b)[1] = *(const bf16x8*)(p + 8);
  return b;
}

// ---------------------------------------------------------------------------
// Prep kernels
// ---------------------------------------------------------------------------

__global__ void k_zero_loss(float* p) {
  if (blockIdx.x == 0 && threadIdx.x == 0) p[0] = 0.0f;
}

// src is [N][K] fp32 (e.g. w_ih [2048][160]); dst is bf16 fragment-major:
// dst[(((kc*NT)+nt)*32 + lane)*16 + e] = src[nt*16+(lane&15)][kc*32+16*(e>>3)+8*(lane>>4)+(e&7)]
__global__ void k_prep_wfrag(__bf16* __restrict__ dst,
                             const float* __restrict__ src, int N, int K) {
  int i = blockIdx.x * 256 + threadIdx.x;
  if (i >= N * K) return;
  int e = i & 15;
  int lane = (i >> 4) & 31;
  int rest = i >> 9;            // kc*NT + nt
  int NT = N >> 4;
  int nt = rest % NT;
  int kc = rest / NT;
  int k = kc * 32 + ((e >> 3) << 4) + ((lane >> 4) << 3) + (e & 7);
  int n = nt * 16 + (lane & 15);
  dst[i] = (__bf16)src[(size_t)n * K + k];
}

// conv weight [co][ci][tap] (ci=128,tap=3) -> fragment-major with
// K index = tap*128+ci (Ktot=384, 12 chunks), N = co (8 tiles)
__global__ void k_prep_convw_frag(__bf16* __restrict__ dst,
                                  const float* __restrict__ w) {
  int i = blockIdx.x * 256 + threadIdx.x;   // 384*128
  if (i >= 384 * 128) return;
  int e = i & 15;
  int lane = (i >> 4) & 31;
  int rest = i >> 9;
  int nt = rest & 7;            // NT = 8
  int kc = rest >> 3;
  int k = kc * 32 + ((e >> 3) << 4) + ((lane >> 4) << 3) + (e & 7);
  int n = nt * 16 + (lane & 15);
  int tap = k >> 7, ci = k & 127;
  dst[i] = (__bf16)w[((size_t)n * 128 + ci) * 3 + tap];
}

// dst[k][n] = src[n][k] (fp32, for the small regression-head weights)
__global__ void k_transpose_f32(float* __restrict__ dst,
                                const float* __restrict__ src, int n, int k) {
  int i = blockIdx.x * 256 + threadIdx.x;
  if (i >= n * k) return;
  int kk = i / n, nn = i - kk * n;
  dst[(size_t)kk * n + nn] = src[(size_t)nn * k + kk];
}

// gsum[j] = sum_f gamma_w[j][f]; weff[co*3+tap] = sum_ci w1_0[co][ci][tap];
// wdsum[co] = sum_ci wd0[co][ci]
__global__ void k_prep_small(float* gsum, float* weff, float* wdsum,
                             const float* gamma_w, const float* w1_0,
                             const float* wd0) {
  int tid = threadIdx.x;   // one block of 512
  if (tid < 512) {
    float s = 0.f;
    for (int f = 0; f < 32; f++) s += gamma_w[tid * 32 + f];
    gsum[tid] = s;
  }
  if (tid < 384) {
    int co = tid / 3, tap = tid - co * 3;
    float s = 0.f;
    for (int ci = 0; ci < 32; ci++) s += w1_0[(co * 32 + ci) * 3 + tap];
    weff[tid] = s;
  }
  if (tid < 128) {
    float s = 0.f;
    for (int ci = 0; ci < 32; ci++) s += wd0[tid * 32 + ci];
    wdsum[tid] = s;
  }
}

// inv_denom[t] = 1 / ((32*(B - sum_b mask[b,t]) + 1e-5) * T)
__global__ void k_denom(float* __restrict__ inv_denom, const float* __restrict__ masks) {
  __shared__ float red[256];
  int t = blockIdx.x, tid = threadIdx.x;
  float s = 0.f;
  for (int b = tid; b < B_SZ; b += 256) s += masks[b * T_SZ + t];
  red[tid] = s; __syncthreads();
  for (int st = 128; st > 0; st >>= 1) {
    if (tid < st) red[tid] += red[tid + st];
    __syncthreads();
  }
  if (tid == 0) {
    float denom = 32.0f * ((float)B_SZ - red[0]) + 1e-5f;
    inv_denom[t] = 1.0f / (denom * (float)T_SZ);
  }
}

// ---------------------------------------------------------------------------
// TCN level-1 conv1: ndvi broadcast -> channel-summed K=3 conv (VALU, cheap)
// ---------------------------------------------------------------------------
__global__ void k_conv1_l1(__bf16* __restrict__ out, const float* __restrict__ ndvi,
                           const float* __restrict__ weff, const float* __restrict__ b1) {
  int i = blockIdx.x * 256 + threadIdx.x;   // B*T*128
  if (i >= B_SZ * T_SZ * CCH) return;
  int co = i & 127, bt = i >> 7;
  int t = bt & 255, b = bt >> 8;
  float s = b1[co];
  for (int tap = 0; tap < 3; tap++) {
    int tp = t - (2 - tap);
    if (tp >= 0) s += weff[co * 3 + tap] * ndvi[b * T_SZ + tp];
  }
  out[(size_t)bt * CCH + co] = (__bf16)fmaxf(s, 0.0f);
}

// ---------------------------------------------------------------------------
// Generic dilated-causal-conv (128->128, K=3) as bf16 WMMA GEMM.
// One wave computes (b, 16-t tile, 64-co half): 48 v_wmma per wave.
// mode 0: out = relu(conv + bias)                              (conv1)
// mode 1: out = relu(relu(conv + bias) + resbuf)               (conv2, lvl 2/3)
// mode 2: out = relu(relu(conv + bias) + ndvi*wdsum + bd)      (conv2, lvl 1)
// ---------------------------------------------------------------------------
__global__ void __launch_bounds__(256)
k_conv_wmma(__bf16* __restrict__ out, const __bf16* __restrict__ x,
            const __bf16* __restrict__ wgF, const float* __restrict__ bias,
            const __bf16* __restrict__ resbuf, const float* __restrict__ ndvi,
            const float* __restrict__ wdsum, const float* __restrict__ bd,
            int dil, int mode) {
  int wave = (blockIdx.x * 256 + threadIdx.x) >> 5;
  int lane = threadIdx.x & 31;
  int cohalf = wave & 1;
  int tt = (wave >> 1) & 15;
  int b = wave >> 5;
  int t0 = tt * 16;
  int m = lane & 15, half = lane >> 4;

  floatx8 acc[4];
  for (int nt = 0; nt < 4; nt++) acc[nt] = splat8(0.0f);

  for (int tap = 0; tap < 3; tap++) {
    int tp = t0 + m - (2 - tap) * dil;           // causal left pad
    for (int cc = 0; cc < 4; cc++) {
      bf16x16 a = (tp >= 0)
          ? load_a(x + ((size_t)(b * T_SZ + tp) * CCH + cc * 32), half)
          : zero_af();
      int kc = tap * 4 + cc;                     // 12 K-chunks total
      for (int nt = 0; nt < 4; nt++) {
        bf16x16 bb = load_b_frag(wgF, 8, kc, cohalf * 4 + nt, lane);
        acc[nt] = WMMA_BF16(a, bb, acc[nt]);
      }
    }
  }

  int n15 = lane & 15, mb = half * 8;
  for (int nt = 0; nt < 4; nt++) {
    int co = cohalf * 64 + nt * 16 + n15;
    float bv = bias[co];
    for (int vg = 0; vg < 8; vg++) {
      int mm = mb + vg, t = t0 + mm;
      size_t oidx = (size_t)(b * T_SZ + t) * CCH + co;
      float v = fmaxf(acc[nt][vg] + bv, 0.0f);
      if (mode == 1) {
        v = fmaxf(v + (float)resbuf[oidx], 0.0f);
      } else if (mode == 2) {
        v = fmaxf(v + ndvi[b * T_SZ + t] * wdsum[co] + bd[co], 0.0f);
      }
      out[oidx] = (__bf16)v;
    }
  }
}

// ---------------------------------------------------------------------------
// Scan kernel: 64 blocks x 512 threads (16 waves). Block owns 16 batch rows
// for all 256 steps. c resident in registers, h in LDS. Per wave per step:
// 1 M-tile x 8 N-tiles x 21 K-chunks = 168 v_wmma_f32_16x16x32_bf16.
// Wave w owns j-tiles {w, w+16} and the matching i/f/g/o gate tiles, so the
// LSTM cell update is fully in-register. Weights stream from L2 (2.7MB bf16,
// fragment-major -> 1KB coalesced wave loads).
// ---------------------------------------------------------------------------
__global__ void __launch_bounds__(512)
k_scan(float* __restrict__ d_out,              // [0]=loss, 1.. = imps [B][32][T]
       const __bf16* __restrict__ tcn,         // [B][T][128] bf16
       const __bf16* __restrict__ wihF,        // frag-major [5][128][32][16]
       const __bf16* __restrict__ whhF,        // frag-major [16][128][32][16]
       const float* __restrict__ srs,          // [B][32][T]
       const float* __restrict__ masks,        // [B][T]
       const float* __restrict__ deltas,       // [B][T]
       const float* __restrict__ gsum_g, const float* __restrict__ gamma_b,
       const float* __restrict__ b_ih, const float* __restrict__ b_hh,
       const float* __restrict__ rlwT,         // [512][32] fp32 (transposed)
       const float* __restrict__ rlb,
       const float* __restrict__ rrwT,         // [32][32] fp32 (transposed)
       const float* __restrict__ rrb,
       const float* __restrict__ inv_denom) {
  extern __shared__ char smem[];
  float*  h_f    = (float*)smem;            // 16*512
  float*  src_f  = h_f + 16 * 512;          // 16*32
  float*  gsum_s = src_f + 16 * 32;         // 512
  float*  gb_s   = gsum_s + 512;            // 512
  float*  red    = gb_s + 512;              // 512
  __bf16* h_bf   = (__bf16*)(red + 512);    // 16*512
  __bf16* src_bf = h_bf + 16 * 512;         // 16*32

  int tid = threadIdx.x, lane = tid & 31, w = tid >> 5;
  int b0 = blockIdx.x * 16;
  int n15 = lane & 15, mrow = lane & 15, half = lane >> 4;

  for (int i = tid; i < 16 * 512; i += 512) h_f[i] = 0.0f;   // h0 = 0
  if (tid < 512) { gsum_s[tid] = gsum_g[tid]; gb_s[tid] = gamma_b[tid]; }

  float creg[2][8];                                           // c0 = 0
  for (int jt = 0; jt < 2; jt++)
    for (int vg = 0; vg < 8; vg++) creg[jt][vg] = 0.0f;

  float biasv[2][4];
  for (int jt = 0; jt < 2; jt++)
    for (int g = 0; g < 4; g++) {
      int n = g * 512 + (w + jt * 16) * 16 + n15;
      biasv[jt][g] = b_ih[n] + b_hh[n];
    }

  int m2 = tid >> 5, f2 = tid & 31;   // (batch row, feature) for elementwise
  float lossAcc = 0.0f;
  __syncthreads();

  for (int t = 0; t < T_SZ; t++) {
    // ---- phase 1: gamma decay, h_f *= gamma, emit bf16 copy ----
    // thread owns fixed batch row m2 -> delta loaded once
    {
      float delta = deltas[(b0 + m2) * T_SZ + t];
      for (int k = 0; k < 16; k++) {
        int j = f2 + k * 32;        // f2 == lane&31 -> conflict-free LDS
        int idx = m2 * 512 + j;
        float gm = expf(-fmaxf(delta * gsum_s[j] + gb_s[j], 0.0f));
        float hv = h_f[idx] * gm;
        h_f[idx] = hv;
        h_bf[idx] = (__bf16)hv;
      }
    }
    __syncthreads();

    // ---- phase 2: sr_h head, loss part 1, sr_c ----
    float sr_y, inv;
    {
      float s = rlb[f2];
      const float* hr = h_f + m2 * 512;          // LDS broadcast across wave
      for (int j = 0; j < 512; j++) s += hr[j] * rlwT[j * 32 + f2];  // coalesced
      sr_y = srs[((size_t)(b0 + m2) * 32 + f2) * T_SZ + t];
      float mk = masks[(b0 + m2) * T_SZ + t];
      inv = 1.0f - mk;
      lossAcc += fabsf(sr_y - s) * inv * inv_denom[t];
      float sc = sr_y * inv + mk * s;
      src_f[m2 * 32 + f2] = sc;
      src_bf[m2 * 32 + f2] = (__bf16)sc;
    }
    __syncthreads();

    // ---- phase 3a: sr_z head, loss part 2, imputation out ----
    {
      float z = rrb[f2];
      const float* sr = src_f + m2 * 32;
      for (int f = 0; f < 32; f++) z += sr[f] * rrwT[f * 32 + f2];   // coalesced
      lossAcc += fabsf(sr_y - z) * inv * inv_denom[t];
      d_out[1 + ((size_t)(b0 + m2) * 32 + f2) * T_SZ + t] = z;
    }

    // ---- phase 3b: gates = [tcn|sr_c|h] @ [w_ihT;w_hhT] via WMMA ----
    floatx8 acc[2][4];
    for (int jt = 0; jt < 2; jt++)
      for (int g = 0; g < 4; g++) acc[jt][g] = splat8(biasv[jt][g]);

    const __bf16* arow_tcn = tcn + ((size_t)(b0 + mrow) * T_SZ + t) * CCH;
    if (t + 1 < T_SZ) __builtin_prefetch(arow_tcn + CCH, 0, 1);  // next-step row

    for (int kc = 0; kc < 4; kc++) {                  // K 0..127: tcn features
      bf16x16 a = load_a(arow_tcn + kc * 32, half);
      for (int jt = 0; jt < 2; jt++)
        for (int g = 0; g < 4; g++) {
          bf16x16 bb = load_b_frag(wihF, 128, kc, g * 32 + (w + jt * 16), lane);
          acc[jt][g] = WMMA_BF16(a, bb, acc[jt][g]);
        }
    }
    {                                                 // K 128..159: sr_c (LDS)
      bf16x16 a = load_a(src_bf + mrow * 32, half);
      for (int jt = 0; jt < 2; jt++)
        for (int g = 0; g < 4; g++) {
          bf16x16 bb = load_b_frag(wihF, 128, 4, g * 32 + (w + jt * 16), lane);
          acc[jt][g] = WMMA_BF16(a, bb, acc[jt][g]);
        }
    }
    for (int kc = 0; kc < 16; kc++) {                 // recurrent: h (LDS)
      bf16x16 a = load_a(h_bf + mrow * 512 + kc * 32, half);
      for (int jt = 0; jt < 2; jt++)
        for (int g = 0; g < 4; g++) {
          bf16x16 bb = load_b_frag(whhF, 128, kc, g * 32 + (w + jt * 16), lane);
          acc[jt][g] = WMMA_BF16(a, bb, acc[jt][g]);
        }
    }

    // ---- phase 4: in-register LSTM cell update ----
    for (int jt = 0; jt < 2; jt++) {
      for (int vg = 0; vg < 8; vg++) {
        float iv = sigf(acc[jt][0][vg]);
        float fv = sigf(acc[jt][1][vg]);
        float gv = tanhf(acc[jt][2][vg]);
        float ov = sigf(acc[jt][3][vg]);
        float c = fv * creg[jt][vg] + iv * gv;
        creg[jt][vg] = c;
        float hv = ov * tanhf(c);
        int mm = half * 8 + vg;
        int j = (w + jt * 16) * 16 + n15;
        h_f[mm * 512 + j] = hv;
      }
    }
    __syncthreads();
  }

  // ---- loss reduction + single atomic per block ----
  red[tid] = lossAcc; __syncthreads();
  for (int st = 256; st > 0; st >>= 1) {
    if (tid < st) red[tid] += red[tid + st];
    __syncthreads();
  }
  if (tid == 0) atomicAdd(d_out, red[0]);
}

// ---------------------------------------------------------------------------
// Host launcher
// ---------------------------------------------------------------------------
extern "C" void kernel_launch(void* const* d_in, const int* in_sizes, int n_in,
                              void* d_out, int out_size, void* d_ws, size_t ws_size,
                              hipStream_t stream) {
  (void)in_sizes; (void)n_in; (void)out_size; (void)ws_size;
  // setup_inputs() flat order (None leaves dropped in tcn_params):
  const float* srs     = (const float*)d_in[0];
  const float* masks   = (const float*)d_in[1];
  const float* deltas  = (const float*)d_in[2];
  const float* ndvi    = (const float*)d_in[3];
  const float* w1_0    = (const float*)d_in[4];
  const float* b1_0    = (const float*)d_in[5];
  const float* w2_0    = (const float*)d_in[6];
  const float* b2_0    = (const float*)d_in[7];
  const float* wd_0    = (const float*)d_in[8];
  const float* bd_0    = (const float*)d_in[9];
  const float* w1_1    = (const float*)d_in[10];
  const float* b1_1    = (const float*)d_in[11];
  const float* w2_1    = (const float*)d_in[12];
  const float* b2_1    = (const float*)d_in[13];
  const float* w1_2    = (const float*)d_in[14];
  const float* b1_2    = (const float*)d_in[15];
  const float* w2_2    = (const float*)d_in[16];
  const float* b2_2    = (const float*)d_in[17];
  const float* gamma_w = (const float*)d_in[18];
  const float* gamma_b = (const float*)d_in[19];
  const float* w_ih    = (const float*)d_in[20];
  const float* w_hh    = (const float*)d_in[21];
  const float* b_ih    = (const float*)d_in[22];
  const float* b_hh    = (const float*)d_in[23];
  const float* rlw     = (const float*)d_in[24];
  const float* rlb     = (const float*)d_in[25];
  const float* rrw     = (const float*)d_in[26];
  const float* rrb     = (const float*)d_in[27];
  float* out = (float*)d_out;

  // workspace carve
  char* ws = (char*)d_ws;
  const size_t BUF = (size_t)B_SZ * T_SZ * CCH * sizeof(__bf16);  // 64 MB
  size_t off = 0;
  auto take = [&](size_t bytes) { void* p = ws + off; off = (off + bytes + 255) & ~(size_t)255; return p; };
  __bf16* buf0   = (__bf16*)take(BUF);
  __bf16* buf1   = (__bf16*)take(BUF);
  __bf16* buf2   = (__bf16*)take(BUF);
  __bf16* wihF   = (__bf16*)take((size_t)KIH  * NGATE * sizeof(__bf16));
  __bf16* whhF   = (__bf16*)take((size_t)HIDN * NGATE * sizeof(__bf16));
  __bf16* wg0    = (__bf16*)take(384 * 128 * sizeof(__bf16));
  __bf16* wg1    = (__bf16*)take(384 * 128 * sizeof(__bf16));
  __bf16* wg2    = (__bf16*)take(384 * 128 * sizeof(__bf16));
  __bf16* wg3    = (__bf16*)take(384 * 128 * sizeof(__bf16));
  __bf16* wg4    = (__bf16*)take(384 * 128 * sizeof(__bf16));
  float*  gsum   = (float*)take(512 * sizeof(float));
  float*  weff   = (float*)take(384 * sizeof(float));
  float*  wdsum  = (float*)take(128 * sizeof(float));
  float*  invden = (float*)take(256 * sizeof(float));
  float*  rlwT   = (float*)take(512 * 32 * sizeof(float));
  float*  rrwT   = (float*)take(32 * 32 * sizeof(float));

  // --- prep ---
  k_zero_loss<<<1, 64, 0, stream>>>(out);
  k_prep_small<<<1, 512, 0, stream>>>(gsum, weff, wdsum, gamma_w, w1_0, wd_0);
  k_prep_wfrag<<<(NGATE * KIH + 255) / 256, 256, 0, stream>>>(wihF, w_ih, NGATE, KIH);
  k_prep_wfrag<<<(NGATE * HIDN + 255) / 256, 256, 0, stream>>>(whhF, w_hh, NGATE, HIDN);
  k_prep_convw_frag<<<(384 * 128 + 255) / 256, 256, 0, stream>>>(wg0, w2_0);
  k_prep_convw_frag<<<(384 * 128 + 255) / 256, 256, 0, stream>>>(wg1, w1_1);
  k_prep_convw_frag<<<(384 * 128 + 255) / 256, 256, 0, stream>>>(wg2, w2_1);
  k_prep_convw_frag<<<(384 * 128 + 255) / 256, 256, 0, stream>>>(wg3, w1_2);
  k_prep_convw_frag<<<(384 * 128 + 255) / 256, 256, 0, stream>>>(wg4, w2_2);
  k_transpose_f32<<<(32 * 512 + 255) / 256, 256, 0, stream>>>(rlwT, rlw, 32, 512);
  k_transpose_f32<<<(32 * 32 + 255) / 256, 256, 0, stream>>>(rrwT, rrw, 32, 32);
  k_denom<<<T_SZ, 256, 0, stream>>>(invden, masks);

  // --- TCN ---
  int nconv = (B_SZ * T_SZ * CCH + 255) / 256;
  k_conv1_l1<<<nconv, 256, 0, stream>>>(buf0, ndvi, weff, b1_0);
  int cblocks = (B_SZ * (T_SZ / 16) * 2) / 8;   // 4096 blocks, 8 waves each
  // L1 conv2 (dil=1), residual = wd(ndvi)+bd
  k_conv_wmma<<<cblocks, 256, 0, stream>>>(buf1, buf0, wg0, b2_0, nullptr,
                                           ndvi, wdsum, bd_0, 1, 2);
  // L2 conv1 (dil=2)
  k_conv_wmma<<<cblocks, 256, 0, stream>>>(buf0, buf1, wg1, b1_1, nullptr,
                                           nullptr, nullptr, nullptr, 2, 0);
  // L2 conv2 (dil=2), residual = L1 output
  k_conv_wmma<<<cblocks, 256, 0, stream>>>(buf2, buf0, wg2, b2_1, buf1,
                                           nullptr, nullptr, nullptr, 2, 1);
  // L3 conv1 (dil=4)
  k_conv_wmma<<<cblocks, 256, 0, stream>>>(buf0, buf2, wg3, b1_2, nullptr,
                                           nullptr, nullptr, nullptr, 4, 0);
  // L3 conv2 (dil=4), residual = L2 output -> tcn final in buf1
  k_conv_wmma<<<cblocks, 256, 0, stream>>>(buf1, buf0, wg4, b2_2, buf2,
                                           nullptr, nullptr, nullptr, 4, 1);

  // --- LSTM scan ---
  size_t smem = (16 * 512 + 16 * 32 + 512 + 512 + 512) * sizeof(float) +
                (16 * 512 + 16 * 32) * sizeof(__bf16);   // ~58 KB
  k_scan<<<B_SZ / 16, 512, smem, stream>>>(out, buf1, wihF, whhF, srs, masks,
                                           deltas, gsum, gamma_b, b_ih, b_hh,
                                           rlwT, rlb, rrwT, rrb, invden);
}
